// VGRNN_85615878078814
// MI455X (gfx1250) — matrix operand order repaired and checked by reading, not compiled, not run on hardware
//
#include <hip/hip_runtime.h>

// ---------------------------------------------------------------------------
// VGRNN forward for MI455X (gfx1250, wave32).
// - All GEMMs (dense MLPs, GCN linear stages, fused z·z^T + BCE decode) run on
//   v_wmma_f32_16x16x32_bf16 (bf16 inputs, f32 accumulate).
// - GEMM B-panels are staged into LDS with global_load_async_to_lds_b128
//   (ASYNCcnt) + s_wait_asynccnt.
// - The BCE kernel's z column-tile is staged via the Tensor Data Mover
//   (tensor_load_to_lds, TENSORcnt) with a hand-built D# descriptor.
// - GCN segment-sum aggregation uses atomic scatter over the edge list.
// ---------------------------------------------------------------------------

typedef __attribute__((ext_vector_type(16))) __bf16 v16bf;
typedef __attribute__((ext_vector_type(8)))  float  v8f;
typedef __attribute__((ext_vector_type(4)))  unsigned int v4u;
typedef __attribute__((ext_vector_type(8)))  int v8i;
typedef __attribute__((ext_vector_type(4)))  int v4i;

constexpr int NB = 4096;   // nodes
constexpr int TT = 4;      // timesteps
constexpr int EE = 65536;  // edges
constexpr int XDc = 256;
constexpr int HDc = 256;
constexpr int ZDc = 64;
constexpr float EPSV = 1e-10f;

__device__ __forceinline__ float softplus_f(float x) {
  return x > 20.f ? x : log1pf(expf(x));
}
__device__ __forceinline__ float sigmoid_f(float x) {
  return 1.f / (1.f + expf(-x));
}
__device__ __forceinline__ float apply_act(float v, int act) {
  switch (act) {
    case 1: return v > 0.f ? v : 0.f;   // relu
    case 2: return softplus_f(v);       // softplus
    case 3: return sigmoid_f(v);        // sigmoid
    case 4: return tanhf(v);            // tanh
    default: return v;
  }
}

// Issue one per-lane async LDS load of 16 bytes: LDS[lds_off] = MEM[gp .. gp+15]
__device__ __forceinline__ void async_copy_b128(const void* gp, const void* lds_ptr) {
  unsigned lds_off = (unsigned)(unsigned long long)lds_ptr;  // LDS address = low 32 bits of the pointer
  asm volatile("global_load_async_to_lds_b128 %0, %1, off"
               :: "v"(lds_off), "v"(gp)
               : "memory");
}
__device__ __forceinline__ void async_wait_all() {
  asm volatile("s_wait_asynccnt 0x0" ::: "memory");
}

// TDM: DMA a 2D tile (tile_d1 rows x tile_d0 elems, f32) from a row-major
// tensor (row length = stride0 elems, tensor_d1 rows total) into LDS.
// D# bit layout per CDNA5 ISA ch.8 (group0 128b, group1 256b; groups 2/3 zero
// for <=2D tensors; count=1, type=2).
__device__ __forceinline__ void tdm_load_tile_f32(const float* gbase, void* lds_ptr,
                                                  unsigned tensor_d0, unsigned tensor_d1,
                                                  unsigned tile_d0, unsigned tile_d1,
                                                  unsigned stride0) {
  const unsigned long long gaddr = (unsigned long long)(const void*)gbase;
  const unsigned lds_addr = (unsigned)(unsigned long long)lds_ptr;
  v4u g0;
  g0[0] = 1u;                                                  // count=1 (valid)
  g0[1] = lds_addr;                                            // lds_addr [63:32]
  g0[2] = (unsigned)(gaddr & 0xFFFFFFFFu);                     // global_addr lo
  g0[3] = (unsigned)((gaddr >> 32) & 0x1FFFFFFu) | (2u << 30); // addr hi | type=2
  v8i g1;
  g1[0] = (int)(2u << 16);                                     // data_size=4B
  g1[1] = (int)((tensor_d0 & 0xFFFFu) << 16);                  // tensor_dim0 lo16
  g1[2] = (int)(((tensor_d0 >> 16) & 0xFFFFu) | ((tensor_d1 & 0xFFFFu) << 16));
  g1[3] = (int)(((tensor_d1 >> 16) & 0xFFFFu) | ((tile_d0 & 0xFFFFu) << 16));
  g1[4] = (int)(tile_d1 & 0xFFFFu);                            // tile_dim1; tile_dim2=0
  g1[5] = (int)stride0;                                        // tensor_dim0_stride lo32
  g1[6] = 0;                                                   // stride hi | dim1_stride lo
  g1[7] = 0;
  const v4i gz4 = {0, 0, 0, 0};
  const v8i gz8 = {0, 0, 0, 0, 0, 0, 0, 0};
  __builtin_amdgcn_tensor_load_to_lds(g0, g1, gz4, gz4, gz8, 0);
  __builtin_amdgcn_s_wait_tensorcnt(0);
}

// ---------------------------------------------------------------------------
// WMMA GEMM: Y = act( (accum ? Y : 0) + X[M,K] @ W[K,NC] + bias )
// Block = 128 threads (4 waves); all waves share output-column tile nt, so the
// W column-panel (K x 16 f32) is async-staged into LDS once per block.
// One 16x16 output tile per wave (m-direction); K templated -> fully unrolled.
// A-fragment (16x32 bf16): lane<16 -> M=lane, K 0..15 contiguous in v0..v7;
// lane>=16 -> same rows, K offset +16.
// B-fragment (32x16 bf16): lane%16 = output col, K half = 16*(lane>>4).
// C/D (16x16 f32): element r of the v8f = row (r + 8*(lane>>4)), col lane%16.
// ---------------------------------------------------------------------------
template <int K>
__global__ void wmma_gemm_kernel(const float* __restrict__ X,
                                 const float* __restrict__ W,
                                 const float* __restrict__ bias,
                                 float* __restrict__ Y,
                                 int NC, int act, int accum) {
  __shared__ float wtile[K * 16];   // W[k, colbase .. colbase+15]

  const int lane = threadIdx.x & 31;
  const int wave = threadIdx.x >> 5;
  const int half = lane >> 4;
  const int l16  = lane & 15;
  const int mt = blockIdx.y * 4 + wave;
  const int nt = blockIdx.x;
  const int rowg = mt * 16 + l16;
  const int colg = nt * 16 + l16;
  const int colbase = nt * 16;

  // ---- async stage of the shared W panel into LDS (K*4 b128 chunks) ----
#pragma unroll 1
  for (int idx = threadIdx.x; idx < K * 4; idx += 128) {
    const int r  = idx >> 2;
    const int c4 = (idx & 3) * 4;
    async_copy_b128(W + (size_t)r * NC + colbase + c4, &wtile[r * 16 + c4]);
  }
  async_wait_all();
  __syncthreads();

  v8f acc = {};
  if (accum) {
#pragma unroll
    for (int r = 0; r < 8; ++r)
      acc[r] = Y[(size_t)(mt * 16 + r + half * 8) * NC + colg];
  }

  const float* xr = X + (size_t)rowg * K;
#pragma unroll
  for (int k0 = 0; k0 < K; k0 += 32) {
    v16bf a, b;
    const int klo = k0 + half * 16;
#pragma unroll
    for (int v = 0; v < 4; ++v) {
      a[2 * v]         = (__bf16)xr[klo + 2 * v];
      a[2 * v + 1]     = (__bf16)xr[klo + 2 * v + 1];
      a[8 + 2 * v]     = (__bf16)xr[klo + 8 + 2 * v];
      a[8 + 2 * v + 1] = (__bf16)xr[klo + 8 + 2 * v + 1];
    }
    const int kb = k0 + half * 16;
#pragma unroll
    for (int v = 0; v < 8; ++v) {
      b[2 * v]     = (__bf16)wtile[(kb + 2 * v) * 16 + l16];
      b[2 * v + 1] = (__bf16)wtile[(kb + 2 * v + 1) * 16 + l16];
    }
    acc = __builtin_amdgcn_wmma_f32_16x16x32_bf16(false, a, false, b,
                                                  (short)0, acc, false, false);
  }

  const float bv = bias ? bias[colg] : 0.f;
#pragma unroll
  for (int r = 0; r < 8; ++r) {
    float v = acc[r] + bv;
    Y[(size_t)(mt * 16 + r + half * 8) * NC + colg] = apply_act(v, act);
  }
}

// ---------------------------------------------------------------------------
// Fused dec = sigmoid(z z^T) + weighted-BCE reduction against adj[t].
// One 16x16 tile of the 4096x4096 logit matrix per wave, K=64 (2 WMMA ops).
// The z column-tile (shared by all waves in the block) is DMA'd to LDS by the
// Tensor Data Mover. The NxN matrix is never materialized.
// ---------------------------------------------------------------------------
__global__ void bce_tiles_kernel(const float* __restrict__ z,
                                 const float* __restrict__ adj,
                                 const float* __restrict__ tsum_p,
                                 float* __restrict__ bce_sum) {
  __shared__ float ztile[16 * ZDc];   // z[jt*16 + r][0..63]
  __shared__ float red[128];

  const int lane = threadIdx.x & 31;
  const int wave = threadIdx.x >> 5;
  const int half = lane >> 4;
  const int l16  = lane & 15;
  const int it = blockIdx.y * 4 + wave;
  const int jt = blockIdx.x;
  const int rowg = it * 16 + l16;
  const int colg = jt * 16 + l16;

  // ---- TDM stage of the shared z column tile (16 rows x 64 f32) ----
  if (wave == 0) {
    tdm_load_tile_f32(z + (size_t)jt * 16 * ZDc, ztile,
                      /*tensor_d0=*/ZDc, /*tensor_d1=*/NB,
                      /*tile_d0=*/ZDc, /*tile_d1=*/16,
                      /*stride0=*/ZDc);
  }
  __syncthreads();

  const float tsum = *tsum_p;
  const float nn = (float)NB * (float)NB;
  const float posw = (nn - tsum) / tsum;

  __builtin_prefetch(adj + (size_t)rowg * NB + colg, 0, 1);

  v8f acc = {};
#pragma unroll
  for (int k0 = 0; k0 < ZDc; k0 += 32) {
    v16bf a, b;
    const float* zr = z + (size_t)rowg * ZDc + k0 + half * 16;
#pragma unroll
    for (int v = 0; v < 4; ++v) {
      a[2 * v]         = (__bf16)zr[2 * v];
      a[2 * v + 1]     = (__bf16)zr[2 * v + 1];
      a[8 + 2 * v]     = (__bf16)zr[8 + 2 * v];
      a[8 + 2 * v + 1] = (__bf16)zr[8 + 2 * v + 1];
    }
    // B[k, col] = z[col, k] from the LDS tile
    const float* zc = &ztile[l16 * ZDc + k0 + half * 16];
#pragma unroll
    for (int v = 0; v < 8; ++v) {
      b[2 * v]     = (__bf16)zc[2 * v];
      b[2 * v + 1] = (__bf16)zc[2 * v + 1];
    }
    acc = __builtin_amdgcn_wmma_f32_16x16x32_bf16(false, a, false, b,
                                                  (short)0, acc, false, false);
  }

  float local = 0.f;
#pragma unroll
  for (int r = 0; r < 8; ++r) {
    const int orow = it * 16 + r + half * 8;
    const float dec = sigmoid_f(acc[r]);   // faithful quirk: BCE on sigmoid output
    const float av = adj[(size_t)orow * NB + colg];
    local += posw * av * softplus_f(-dec) + (1.f - av) * softplus_f(dec);
  }

  red[threadIdx.x] = local;
  __syncthreads();
  for (int s = 64; s > 0; s >>= 1) {
    if (threadIdx.x < s) red[threadIdx.x] += red[threadIdx.x + s];
    __syncthreads();
  }
  if (threadIdx.x == 0) atomicAdd(bce_sum, red[0]);
}

// --------------------------- small helper kernels ---------------------------
__global__ void set_const_kernel(float* __restrict__ p, float v, int n) {
  for (int i = blockIdx.x * blockDim.x + threadIdx.x; i < n; i += gridDim.x * blockDim.x)
    p[i] = v;
}
__global__ void copy_kernel(const float* __restrict__ a, float* __restrict__ b, int n) {
  for (int i = blockIdx.x * blockDim.x + threadIdx.x; i < n; i += gridDim.x * blockDim.x)
    b[i] = a[i];
}
__global__ void deg_count_kernel(const int* __restrict__ dst, float* __restrict__ deg, int e) {
  for (int i = blockIdx.x * blockDim.x + threadIdx.x; i < e; i += gridDim.x * blockDim.x)
    atomicAdd(&deg[dst[i]], 1.f);
}
__global__ void make_dinv_kernel(const float* __restrict__ deg, float* __restrict__ dinv, int n) {
  for (int i = blockIdx.x * blockDim.x + threadIdx.x; i < n; i += gridDim.x * blockDim.x)
    dinv[i] = rsqrtf(deg[i]);
}
__global__ void gcn_scatter_kernel(const float* __restrict__ hlin,
                                   const int* __restrict__ src,
                                   const int* __restrict__ dst,
                                   const float* __restrict__ dinv,
                                   float* __restrict__ agg, int F) {
  const int total = EE * F;
  for (int i = blockIdx.x * blockDim.x + threadIdx.x; i < total; i += gridDim.x * blockDim.x) {
    const int e = i / F;
    const int f = i - e * F;
    const int s = src[e], d = dst[e];
    const float nrm = dinv[s] * dinv[d];
    atomicAdd(&agg[(size_t)d * F + f], hlin[(size_t)s * F + f] * nrm);
  }
}
// out = act(agg + dinv^2 * hlin + bias + other)
__global__ void gcn_finalize_kernel(const float* __restrict__ agg,
                                    const float* __restrict__ hlin,
                                    const float* __restrict__ dinv,
                                    const float* __restrict__ bias,
                                    const float* __restrict__ other,
                                    float* __restrict__ out, int F, int act) {
  const int n = NB * F;
  for (int i = blockIdx.x * blockDim.x + threadIdx.x; i < n; i += gridDim.x * blockDim.x) {
    const int node = i / F, f = i - node * F;
    const float di = dinv[node];
    float v = agg[i] + di * di * hlin[i];
    if (bias)  v += bias[f];
    if (other) v += other[i];
    out[i] = apply_act(v, act);
  }
}
__global__ void z_kld_kernel(const float* __restrict__ enc_mean,
                             const float* __restrict__ enc_std,
                             const float* __restrict__ prior_mean,
                             const float* __restrict__ prior_std,
                             const float* __restrict__ eps,
                             float* __restrict__ z,
                             float* __restrict__ kld_out, int n, float scale) {
  __shared__ float red[256];
  float local = 0.f;
  for (int i = blockIdx.x * blockDim.x + threadIdx.x; i < n; i += gridDim.x * blockDim.x) {
    const float m1 = enc_mean[i], s1 = enc_std[i] + EPSV;
    const float m2 = prior_mean[i], s2 = prior_std[i] + EPSV;
    z[i] = eps[i] * enc_std[i] + enc_mean[i];
    const float dm = m1 - m2;
    local += 2.f * logf(s2) - 2.f * logf(s1) - (s1 * s1 + dm * dm) / (s2 * s2) + 1.f;
  }
  red[threadIdx.x] = local;
  __syncthreads();
  for (int s = 128; s > 0; s >>= 1) {
    if (threadIdx.x < s) red[threadIdx.x] += red[threadIdx.x + s];
    __syncthreads();
  }
  if (threadIdx.x == 0) atomicAdd(kld_out, red[0] * scale);
}
__global__ void reduce_sum_kernel(const float* __restrict__ a, float* __restrict__ out, int n) {
  __shared__ float red[256];
  float local = 0.f;
  for (int i = blockIdx.x * blockDim.x + threadIdx.x; i < n; i += gridDim.x * blockDim.x)
    local += a[i];
  red[threadIdx.x] = local;
  __syncthreads();
  for (int s = 128; s > 0; s >>= 1) {
    if (threadIdx.x < s) red[threadIdx.x] += red[threadIdx.x + s];
    __syncthreads();
  }
  if (threadIdx.x == 0) atomicAdd(out, red[0]);
}
__global__ void nll_finalize_kernel(const float* __restrict__ tsum_p,
                                    const float* __restrict__ bsum_p,
                                    float* __restrict__ nll_out) {
  if (threadIdx.x == 0 && blockIdx.x == 0) {
    const float tsum = *tsum_p;
    const float nn = (float)NB * (float)NB;
    // norm * mean(bce) = [nn/((nn-tsum)*2)] * bsum/nn = bsum / (2*(nn-tsum))
    *nll_out += *bsum_p / (2.f * (nn - tsum));
  }
}
__global__ void ew_mul_kernel(const float* __restrict__ a, const float* __restrict__ b,
                              float* __restrict__ out, int n) {
  for (int i = blockIdx.x * blockDim.x + threadIdx.x; i < n; i += gridDim.x * blockDim.x)
    out[i] = a[i] * b[i];
}
__global__ void gru_combine_kernel(const float* __restrict__ zg, const float* __restrict__ h,
                                   const float* __restrict__ ht, float* __restrict__ hn, int n) {
  for (int i = blockIdx.x * blockDim.x + threadIdx.x; i < n; i += gridDim.x * blockDim.x) {
    const float zgv = zg[i];
    hn[i] = zgv * h[i] + (1.f - zgv) * ht[i];
  }
}

// ---------------------------------------------------------------------------
extern "C" void kernel_launch(void* const* d_in, const int* in_sizes, int n_in,
                              void* d_out, int out_size, void* d_ws, size_t ws_size,
                              hipStream_t stream) {
  (void)in_sizes; (void)n_in; (void)out_size; (void)ws_size;

  const float* x        = (const float*)d_in[0];
  const int*   edge_idx = (const int*)d_in[1];
  const float* adj      = (const float*)d_in[2];
  const float* eps      = (const float*)d_in[3];
  const float* h0       = (const float*)d_in[4];
  const float* phi_x_w  = (const float*)d_in[5];
  const float* phi_x_b  = (const float*)d_in[6];
  const float* phi_z_w  = (const float*)d_in[7];
  const float* phi_z_b  = (const float*)d_in[8];
  const float* enc_w    = (const float*)d_in[9];
  const float* enc_b    = (const float*)d_in[10];
  const float* enc_mean_w = (const float*)d_in[11];
  const float* enc_mean_b = (const float*)d_in[12];
  const float* enc_std_w  = (const float*)d_in[13];
  const float* enc_std_b  = (const float*)d_in[14];
  const float* prior_w    = (const float*)d_in[15];
  const float* prior_b    = (const float*)d_in[16];
  const float* prior_mean_w = (const float*)d_in[17];
  const float* prior_mean_b = (const float*)d_in[18];
  const float* prior_std_w  = (const float*)d_in[19];
  const float* prior_std_b  = (const float*)d_in[20];
  const float* gru_xz_w = (const float*)d_in[21];
  const float* gru_hz_w = (const float*)d_in[22];
  const float* gru_xr_w = (const float*)d_in[23];
  const float* gru_hr_w = (const float*)d_in[24];
  const float* gru_xh_w = (const float*)d_in[25];
  const float* gru_hh_w = (const float*)d_in[26];

  float* out = (float*)d_out;
  float* kld_out = out + 0;
  float* nll_out = out + 1;
  float* enc_means   = out + 2;
  float* prior_means = out + 2 + (size_t)TT * NB * ZDc;
  float* h_out       = out + 2 + (size_t)2 * TT * NB * ZDc;

  // ---- workspace partition (floats) ----
  float* w = (float*)d_ws;
  auto alloc = [&](size_t n) { float* p = w; w += n; return p; };
  float* hA    = alloc((size_t)NB * HDc);
  float* hB    = alloc((size_t)NB * HDc);
  float* phix  = alloc((size_t)NB * HDc);
  float* phiz  = alloc((size_t)NB * HDc);
  float* lin   = alloc((size_t)NB * HDc);
  float* lin2  = alloc((size_t)NB * HDc);
  float* agg   = alloc((size_t)NB * HDc);
  float* enc_t = alloc((size_t)NB * HDc);
  float* priort= alloc((size_t)NB * HDc);
  float* az    = alloc((size_t)NB * HDc);   // reused for az / ar / axh
  float* zg    = alloc((size_t)NB * HDc);
  float* rg    = alloc((size_t)NB * HDc);
  float* htl   = alloc((size_t)NB * HDc);
  float* rh    = alloc((size_t)NB * HDc);
  float* enc_stdb   = alloc((size_t)NB * ZDc);
  float* prior_stdb = alloc((size_t)NB * ZDc);
  float* zbuf  = alloc((size_t)NB * ZDc);
  float* deg   = alloc(NB);
  float* dinv  = alloc(NB);
  float* tsum  = alloc(1);
  float* bsum  = alloc(1);

  const dim3 ew256(256);
  auto blocks_for = [](int n) { return dim3((n + 255) / 256); };

  auto gemm = [&](const float* X, const float* Wm, const float* B, float* Y,
                  int K, int NC, int act, int accum) {
    const dim3 g(NC / 16, NB / 64);
    if (K == 256)
      wmma_gemm_kernel<256><<<g, 128, 0, stream>>>(X, Wm, B, Y, NC, act, accum);
    else
      wmma_gemm_kernel<64><<<g, 128, 0, stream>>>(X, Wm, B, Y, NC, act, accum);
  };
  auto zero = [&](float* p, int n) {
    set_const_kernel<<<blocks_for(n), ew256, 0, stream>>>(p, 0.f, n);
  };
  auto gcn_agg = [&](const float* hlin, const int* src, const int* dst,
                     const float* bias, const float* other, float* dstbuf, int F, int act) {
    zero(agg, NB * F);
    gcn_scatter_kernel<<<dim3(4096), ew256, 0, stream>>>(hlin, src, dst, dinv, agg, F);
    gcn_finalize_kernel<<<blocks_for(NB * F), ew256, 0, stream>>>(agg, hlin, dinv, bias, other, dstbuf, F, act);
  };

  // init
  zero(out, 2);  // kld, nll accumulators
  copy_kernel<<<blocks_for(NB * HDc), ew256, 0, stream>>>(h0, hA, NB * HDc);

  for (int t = 0; t < TT; ++t) {
    const int*   src   = edge_idx + (size_t)t * 2 * EE;
    const int*   dstp  = src + EE;
    const float* adj_t = adj + (size_t)t * NB * NB;
    const float* eps_t = eps + (size_t)t * NB * ZDc;
    const float* x_t   = x + (size_t)t * NB * XDc;
    float* h  = (t & 1) ? hB : hA;
    float* hn = (t & 1) ? hA : hB;
    float* enc_mean_t   = enc_means + (size_t)t * NB * ZDc;
    float* prior_mean_t = prior_means + (size_t)t * NB * ZDc;

    // degree / normalization for this timestep's edges
    set_const_kernel<<<blocks_for(NB), ew256, 0, stream>>>(deg, 1.f, NB);
    deg_count_kernel<<<dim3(256), ew256, 0, stream>>>(dstp, deg, EE);
    make_dinv_kernel<<<blocks_for(NB), ew256, 0, stream>>>(deg, dinv, NB);

    // phi_x = relu(x_t @ W + b)
    gemm(x_t, phi_x_w, phi_x_b, phix, XDc, HDc, 1, 0);

    // enc_t = relu(gcn(concat[phi_x, h]));  concat handled as split GEMM
    gemm(phix, enc_w,             nullptr, lin, HDc, HDc, 0, 0);
    gemm(h,    enc_w + HDc * HDc, nullptr, lin, HDc, HDc, 0, 1);
    gcn_agg(lin, src, dstp, enc_b, nullptr, enc_t, HDc, 1);

    // enc_mean / enc_std (GCNs)
    gemm(enc_t, enc_mean_w, nullptr, lin, HDc, ZDc, 0, 0);
    gcn_agg(lin, src, dstp, enc_mean_b, nullptr, enc_mean_t, ZDc, 0);
    gemm(enc_t, enc_std_w, nullptr, lin, HDc, ZDc, 0, 0);
    gcn_agg(lin, src, dstp, enc_std_b, nullptr, enc_stdb, ZDc, 2);

    // prior MLPs
    gemm(h, prior_w, prior_b, priort, HDc, HDc, 1, 0);
    gemm(priort, prior_mean_w, prior_mean_b, prior_mean_t, HDc, ZDc, 0, 0);
    gemm(priort, prior_std_w,  prior_std_b,  prior_stdb,   HDc, ZDc, 2, 0);

    // reparameterize + KLD accumulation
    z_kld_kernel<<<dim3(256), ew256, 0, stream>>>(enc_mean_t, enc_stdb, prior_mean_t,
                                                  prior_stdb, eps_t, zbuf, kld_out,
                                                  NB * ZDc, 0.5f / ((float)NB * (float)ZDc));

    // phi_z = relu(z @ W + b)
    gemm(zbuf, phi_z_w, phi_z_b, phiz, ZDc, HDc, 1, 0);

    // NLL: tsum over adj, then fused sigmoid(z z^T) + BCE tiles
    zero(tsum, 1);
    reduce_sum_kernel<<<dim3(4096), ew256, 0, stream>>>(adj_t, tsum, NB * NB);
    zero(bsum, 1);
    bce_tiles_kernel<<<dim3(NB / 16, NB / 64), 128, 0, stream>>>(zbuf, adj_t, tsum, bsum);
    nll_finalize_kernel<<<1, 1, 0, stream>>>(tsum, bsum, nll_out);

    // ---- graph-GRU step (gin = concat[phi_x, phi_z] via split GEMMs) ----
    // z gate
    gemm(phix, gru_xz_w,             nullptr, lin, HDc, HDc, 0, 0);
    gemm(phiz, gru_xz_w + HDc * HDc, nullptr, lin, HDc, HDc, 0, 1);
    gcn_agg(lin, src, dstp, nullptr, nullptr, az, HDc, 0);
    gemm(h, gru_hz_w, nullptr, lin2, HDc, HDc, 0, 0);
    gcn_agg(lin2, src, dstp, nullptr, az, zg, HDc, 3);
    // r gate
    gemm(phix, gru_xr_w,             nullptr, lin, HDc, HDc, 0, 0);
    gemm(phiz, gru_xr_w + HDc * HDc, nullptr, lin, HDc, HDc, 0, 1);
    gcn_agg(lin, src, dstp, nullptr, nullptr, az, HDc, 0);
    gemm(h, gru_hr_w, nullptr, lin2, HDc, HDc, 0, 0);
    gcn_agg(lin2, src, dstp, nullptr, az, rg, HDc, 3);
    // candidate
    gemm(phix, gru_xh_w,             nullptr, lin, HDc, HDc, 0, 0);
    gemm(phiz, gru_xh_w + HDc * HDc, nullptr, lin, HDc, HDc, 0, 1);
    gcn_agg(lin, src, dstp, nullptr, nullptr, az, HDc, 0);  // axh lives in az
    ew_mul_kernel<<<blocks_for(NB * HDc), ew256, 0, stream>>>(rg, h, rh, NB * HDc);
    gemm(rh, gru_hh_w, nullptr, lin2, HDc, HDc, 0, 0);
    gcn_agg(lin2, src, dstp, nullptr, az, htl, HDc, 4);
    // h_new = zg*h + (1-zg)*h_tilde
    gru_combine_kernel<<<blocks_for(NB * HDc), ew256, 0, stream>>>(zg, h, htl, hn, NB * HDc);
  }

  // final hidden state (after TT=4 steps it sits in hA)
  copy_kernel<<<blocks_for(NB * HDc), ew256, 0, stream>>>(hA, h_out, NB * HDc);
}
